// SwinOnlyNet_82291573392101
// MI455X (gfx1250) — matrix operand order repaired and checked by reading, not compiled, not run on hardware
//
#include <hip/hip_runtime.h>
#include <math.h>

// ---------------------------------------------------------------------------
// Types for CDNA5 WMMA (wave32): V_WMMA_F32_16X16X32_F16
// ---------------------------------------------------------------------------
typedef __attribute__((ext_vector_type(16))) _Float16 v16h;
typedef __attribute__((ext_vector_type(8)))  _Float16 v8h;
typedef __attribute__((ext_vector_type(4)))  _Float16 v4h;
typedef __attribute__((ext_vector_type(8)))  float    v8f;
typedef __attribute__((ext_vector_type(4)))  float    v4f;

__device__ __forceinline__ v8f wmma_f16(v16h a, v16h b, v8f c) {
    return __builtin_amdgcn_wmma_f32_16x16x32_f16(
        false, a, false, b, (short)0, c, false, false);
}

// A-fragment K index for element e (0..15) of lane `lane` (16-bit A 16x32):
//   lanes 0-15:  V0..V3 -> K 0..7,  V4..V7 -> K 16..23
//   lanes 16-31: V0..V3 -> K 8..15, V4..V7 -> K 24..31
__device__ __forceinline__ int a_frag_k(int lane, int e) {
    int hi = (lane >> 4) & 1;
    int base = (e < 8) ? (hi ? 8 : 0) : (hi ? 24 : 16);
    return base + (e & 7);
}

__device__ __forceinline__ float gelu_exact(float x) {
    return 0.5f * x * (1.0f + erff(x * 0.70710678118654752f));
}

// store a float4 either as f32 (b128) or converted to f16 (b64)
__device__ __forceinline__ void store4(float* p, v4f v) { *(v4f*)p = v; }
__device__ __forceinline__ void store4(_Float16* p, v4f v) {
    v4h h;
    #pragma unroll
    for (int j = 0; j < 4; ++j) h[j] = (_Float16)v[j];
    *(v4h*)p = h;
}

// ---------------------------------------------------------------------------
// GEMM: out[M,N] = act(A_f16[M,K] * W_f32[K,N] + bias) (+ resid)
// Invariants: M % 64 == 0, K % 32 == 0, N % 4 == 0.
// 64x64 tile / 128 threads (4 waves). A staged as pure b128 f16 copies
// (no conversion); W converted f32->f16 once per tile. Next-tile register
// prefetch overlaps the WMMA chain; all B fragments loaded before WMMAs.
// ---------------------------------------------------------------------------
#define SA 40   // As row stride (halves), 64 rows
#define SB 72   // Bs row stride (halves), 32 rows

template <typename TO>
__global__ __launch_bounds__(128) void gemm_kernel(
    const _Float16* __restrict__ A, const float* __restrict__ W,
    const float* __restrict__ bias, const float* __restrict__ resid,
    TO* __restrict__ out, int M, int N, int K, int act)
{
    __shared__ __align__(16) _Float16 As[64 * SA];
    __shared__ __align__(16) _Float16 Bs[32 * SB];

    const int tid  = threadIdx.x;
    const int lane = tid & 31;
    const int wave = tid >> 5;
    const int m0   = blockIdx.x * 64;
    const int n0   = blockIdx.y * 64;
    const bool fullN = (n0 + 64 <= N);

    v8f acc[4];
    #pragma unroll
    for (int t = 0; t < 4; ++t)
        #pragma unroll
        for (int r = 0; r < 8; ++r) acc[t][r] = 0.0f;

    const v4f z4 = {0.0f, 0.0f, 0.0f, 0.0f};
    v8h ar[2];   // A: 256 chunks of 8 halves (16B), 2 per thread
    v4f br[4];   // W: 512 chunks of 4 floats, 4 per thread

    #pragma unroll
    for (int i = 0; i < 2; ++i) {
        int idx = tid + i * 128;
        ar[i] = *(const v8h*)(A + (size_t)(m0 + (idx >> 2)) * K + ((idx & 3) << 3));
    }
    #pragma unroll
    for (int i = 0; i < 4; ++i) {
        int idx = tid + i * 128;
        int gn = n0 + ((idx & 15) << 2);
        br[i] = (fullN || gn < N)
              ? *(const v4f*)(W + (size_t)(idx >> 4) * N + gn) : z4;
    }

    const int arow = wave * 16 + (lane & 15);

    for (int k0 = 0; k0 < K; k0 += 32) {
        __syncthreads();   // protect previous iteration's LDS reads
        #pragma unroll
        for (int i = 0; i < 2; ++i) {
            int idx = tid + i * 128;
            *(v8h*)(As + (idx >> 2) * SA + ((idx & 3) << 3)) = ar[i];
        }
        #pragma unroll
        for (int i = 0; i < 4; ++i) {
            int idx = tid + i * 128;
            v4h h;
            #pragma unroll
            for (int j = 0; j < 4; ++j) h[j] = (_Float16)br[i][j];
            *(v4h*)(Bs + (idx >> 4) * SB + ((idx & 15) << 2)) = h;
        }
        __syncthreads();

        // issue next tile's global loads before the WMMAs
        int kn = k0 + 32;
        if (kn < K) {
            #pragma unroll
            for (int i = 0; i < 2; ++i) {
                int idx = tid + i * 128;
                ar[i] = *(const v8h*)(A + (size_t)(m0 + (idx >> 2)) * K + kn + ((idx & 3) << 3));
            }
            #pragma unroll
            for (int i = 0; i < 4; ++i) {
                int idx = tid + i * 128;
                int gn = n0 + ((idx & 15) << 2);
                br[i] = (fullN || gn < N)
                      ? *(const v4f*)(W + (size_t)(kn + (idx >> 4)) * N + gn) : z4;
            }
        }

        v16h a;
        #pragma unroll
        for (int e = 0; e < 16; ++e) a[e] = As[arow * SA + a_frag_k(lane, e)];
        v16h bf[4];
        #pragma unroll
        for (int t = 0; t < 4; ++t)
            #pragma unroll
            for (int e = 0; e < 16; ++e) bf[t][e] = Bs[a_frag_k(lane, e) * SB + t * 16 + (lane & 15)];
        #pragma unroll
        for (int t = 0; t < 4; ++t) acc[t] = wmma_f16(a, bf[t], acc[t]);
    }

    const int rbase = m0 + wave * 16 + ((lane >> 4) ? 8 : 0);
    #pragma unroll
    for (int t = 0; t < 4; ++t) {
        int col = n0 + t * 16 + (lane & 15);
        if (col >= N) continue;
        float bv = bias ? bias[col] : 0.0f;
        #pragma unroll
        for (int r = 0; r < 8; ++r) {
            int row = rbase + r;                       // M % 64 == 0: no guard
            float v = acc[t][r] + bv;
            if (act == 1) v = gelu_exact(v);
            if (resid) v += resid[(size_t)row * N + col];
            out[(size_t)row * N + col] = (TO)v;
        }
    }
}

// ---------------------------------------------------------------------------
// Fused window attention (f16 in / f16 out): one block per (window, head).
// N = 49 padded to 64, hd = 32. K stored transposed (kT[d][n]); QK^T B-fragment
// elements of a lane are strided by ST halves per k.
// ---------------------------------------------------------------------------
__device__ __forceinline__ int region_label(int h, int Hs) {
    return (h >= Hs - 3) ? 2 : ((h >= Hs - 7) ? 1 : 0);
}

#define SQ 40   // qs/vs row stride (halves), 64 rows
#define ST 72   // kT row stride (halves), 32 rows
#define SS 72   // Sh row stride (halves), 64 rows
#define SF 65   // Sf row stride (floats), 64 rows

__global__ __launch_bounds__(128) void attn_kernel(
    const _Float16* __restrict__ qkv, const float* __restrict__ rpb,
    _Float16* __restrict__ out, int nWinBatch, int heads, int C,
    int H, int W, int shift)
{
    __shared__ __align__(16) _Float16 qs[64 * SQ];
    __shared__ __align__(16) _Float16 kT[32 * ST];
    __shared__ __align__(16) _Float16 vs[64 * SQ];
    __shared__ __align__(16) float    Sf[64 * SF];
    __shared__ __align__(16) _Float16 Sh[64 * SS];

    const int bid  = blockIdx.x;
    const int head = bid % heads;
    const int win  = bid / heads;
    const int tid  = threadIdx.x;
    const int lane = tid & 31;
    const int wave = tid >> 5;
    const int C3   = 3 * C;
    const float scale = 0.17677669529663687f;  // 32^-0.5

    // load q (scaled), k (transposed), v; pad rows 49..63 with zero
    #pragma unroll
    for (int i = 0; i < 2; ++i) {
        int idx = tid + i * 128;                 // 256 chunks = 64 rows x 4
        int n = idx >> 2, d8 = (idx & 3) << 3;
        v8h q, k, v;
        #pragma unroll
        for (int j = 0; j < 8; ++j) { q[j] = (_Float16)0.0f; k[j] = q[j]; v[j] = q[j]; }
        if (n < 49) {
            const _Float16* base = qkv + (size_t)(win * 49 + n) * C3 + head * 32 + d8;
            q = *(const v8h*)base;
            k = *(const v8h*)(base + C);
            v = *(const v8h*)(base + 2 * C);
        }
        v8h qh;
        #pragma unroll
        for (int j = 0; j < 8; ++j) qh[j] = (_Float16)((float)q[j] * scale);
        *(v8h*)(qs + n * SQ + d8) = qh;
        *(v8h*)(vs + n * SQ + d8) = v;
        #pragma unroll
        for (int j = 0; j < 8; ++j) kT[(d8 + j) * ST + n] = k[j];
    }
    __syncthreads();

    // S = q * k^T : each wave computes a 16-row strip, 4 col tiles (K=32)
    const int arow = wave * 16 + (lane & 15);
    {
        v16h a;
        #pragma unroll
        for (int e = 0; e < 16; ++e) a[e] = qs[arow * SQ + a_frag_k(lane, e)];
        v16h bf[4];
        #pragma unroll
        for (int t = 0; t < 4; ++t)
            #pragma unroll
            for (int e = 0; e < 16; ++e) bf[t][e] = kT[a_frag_k(lane, e) * ST + t * 16 + (lane & 15)];
        int col0 = lane & 15;
        int rb   = wave * 16 + ((lane >> 4) ? 8 : 0);
        #pragma unroll
        for (int t = 0; t < 4; ++t) {
            v8f c;
            #pragma unroll
            for (int r = 0; r < 8; ++r) c[r] = 0.0f;
            c = wmma_f16(a, bf[t], c);
            #pragma unroll
            for (int r = 0; r < 8; ++r) Sf[(rb + r) * SF + t * 16 + col0] = c[r];
        }
    }
    __syncthreads();

    // softmax over 49 cols with rel-pos bias + shifted-window mask (in place)
    if (tid < 64) {
        int n = tid;
        if (n < 49) {
            int ih = n / 7, iw = n % 7;
            int Wn = W / 7;
            int wb = win % nWinBatch;
            int wh = wb / Wn, ww = wb % Wn;
            int idi = 0;
            if (shift > 0)
                idi = region_label(wh * 7 + ih, H) * 3 + region_label(ww * 7 + iw, W);
            float mx = -1e30f;
            for (int j = 0; j < 49; ++j) {
                int jh = j / 7, jw = j % 7;
                float v = Sf[n * SF + j]
                        + rpb[((ih - jh + 6) * 13 + (iw - jw + 6)) * heads + head];
                if (shift > 0) {
                    int idj = region_label(wh * 7 + jh, H) * 3 + region_label(ww * 7 + jw, W);
                    if (idj != idi) v -= 100.0f;
                }
                Sf[n * SF + j] = v;
                mx = fmaxf(mx, v);
            }
            float s = 0.0f;
            for (int j = 0; j < 49; ++j) {
                float e = __expf(Sf[n * SF + j] - mx);
                s += e;
                Sf[n * SF + j] = e;
            }
            float inv = 1.0f / s;
            for (int j = 0; j < 49; ++j)
                Sh[n * SS + j] = (_Float16)(Sf[n * SF + j] * inv);
            for (int j = 49; j < 64; ++j) Sh[n * SS + j] = (_Float16)0.0f;
        } else {
            for (int j = 0; j < 64; ++j) Sh[n * SS + j] = (_Float16)0.0f;
        }
    }
    __syncthreads();

    // out = S * V  (K=64 -> 2 WMMA K-steps; N=32 -> 2 col tiles)
    v8f acc[2];
    #pragma unroll
    for (int t = 0; t < 2; ++t)
        #pragma unroll
        for (int r = 0; r < 8; ++r) acc[t][r] = 0.0f;

    #pragma unroll
    for (int k0 = 0; k0 < 64; k0 += 32) {
        v16h a;
        #pragma unroll
        for (int e = 0; e < 16; ++e) a[e] = Sh[arow * SS + k0 + a_frag_k(lane, e)];
        v16h bf[2];
        #pragma unroll
        for (int t = 0; t < 2; ++t)
            #pragma unroll
            for (int e = 0; e < 16; ++e) bf[t][e] = vs[(k0 + a_frag_k(lane, e)) * SQ + t * 16 + (lane & 15)];
        #pragma unroll
        for (int t = 0; t < 2; ++t) acc[t] = wmma_f16(a, bf[t], acc[t]);
    }
    const int rb = wave * 16 + ((lane >> 4) ? 8 : 0);
    #pragma unroll
    for (int t = 0; t < 2; ++t) {
        int col = head * 32 + t * 16 + (lane & 15);
        #pragma unroll
        for (int r = 0; r < 8; ++r) {
            int row = rb + r;
            if (row < 49)
                out[(size_t)(win * 49 + row) * C + col] = (_Float16)acc[t][r];
        }
    }
}

// ---------------------------------------------------------------------------
// LayerNorm: one wave32 per row; f32 input, templated output (f16 for the
// GEMM activation stream, f32 only for the patch-embed norm on the residual)
// ---------------------------------------------------------------------------
template <typename TO>
__global__ __launch_bounds__(128) void ln_kernel(
    const float* __restrict__ x, const float* __restrict__ g,
    const float* __restrict__ b, TO* __restrict__ out, int M, int C)
{
    int row  = blockIdx.x * 4 + (threadIdx.x >> 5);
    int lane = threadIdx.x & 31;
    if (row >= M) return;
    const float* xr = x + (size_t)row * C;

    float s = 0.0f;
    for (int c = lane * 4; c < C; c += 128) {
        v4f v = *(const v4f*)(xr + c);
        s += v[0] + v[1] + v[2] + v[3];
    }
    #pragma unroll
    for (int m = 16; m >= 1; m >>= 1) s += __shfl_xor(s, m, 32);
    float mean = s / (float)C;

    float var = 0.0f;
    for (int c = lane * 4; c < C; c += 128) {
        v4f v = *(const v4f*)(xr + c);
        #pragma unroll
        for (int j = 0; j < 4; ++j) { float d = v[j] - mean; var += d * d; }
    }
    #pragma unroll
    for (int m = 16; m >= 1; m >>= 1) var += __shfl_xor(var, m, 32);
    float inv = rsqrtf(var / (float)C + 1e-5f);

    TO* orow = out + (size_t)row * C;
    for (int c = lane * 4; c < C; c += 128) {
        v4f v  = *(const v4f*)(xr + c);
        v4f gg = *(const v4f*)(g + c);
        v4f bb = *(const v4f*)(b + c);
        v4f o;
        #pragma unroll
        for (int j = 0; j < 4; ++j) o[j] = (v[j] - mean) * inv * gg[j] + bb[j];
        store4(orow + c, o);
    }
}

// ---------------------------------------------------------------------------
// Patch embed: 4x4 stride-4 conv 3->96, float4 dot products
// ---------------------------------------------------------------------------
__global__ __launch_bounds__(256) void patch_embed_kernel(
    const float* __restrict__ img, const float* __restrict__ w,
    const float* __restrict__ bias, float* __restrict__ out, int B)
{
    size_t i = (size_t)blockIdx.x * 256 + threadIdx.x;
    size_t total = (size_t)B * 56 * 56 * 96;
    if (i >= total) return;
    int c = (int)(i % 96);
    size_t t = i / 96;
    int px = (int)(t % 56);
    int py = (int)((t / 56) % 56);
    int b  = (int)(t / (56 * 56));
    float s = bias[c];
    #pragma unroll
    for (int ci = 0; ci < 3; ++ci)
        #pragma unroll
        for (int ky = 0; ky < 4; ++ky) {
            v4f iv = *(const v4f*)(img + (((size_t)b * 3 + ci) * 224 + py * 4 + ky) * 224 + px * 4);
            v4f wv = *(const v4f*)(w + ((c * 3 + ci) * 4 + ky) * 4);
            s += iv[0] * wv[0] + iv[1] * wv[1] + iv[2] * wv[2] + iv[3] * wv[3];
        }
    out[i] = s;
}

// ---------------------------------------------------------------------------
// roll(-shift) + window partition, f16 16-byte chunks (C % 8 == 0)
// ---------------------------------------------------------------------------
__global__ __launch_bounds__(256) void wpart_kernel(
    const _Float16* __restrict__ xn, _Float16* __restrict__ xw,
    int B, int H, int W, int C, int shift)
{
    size_t i = (size_t)blockIdx.x * 256 + threadIdx.x;
    int Cq = C >> 3;
    size_t total = (size_t)B * H * W * Cq;
    if (i >= total) return;
    int c = (int)(i % Cq) << 3;
    size_t r = i / Cq;
    int n = (int)(r % 49);
    size_t win = r / 49;
    int Wn = W / 7, nW = (H / 7) * Wn;
    int b  = (int)(win / nW);
    int wb = (int)(win % nW);
    int wh = wb / Wn, ww = wb % Wn;
    int h = (wh * 7 + n / 7 + shift) % H;
    int w = (ww * 7 + n % 7 + shift) % W;
    *(v8h*)(xw + i * 8) =
        *(const v8h*)(xn + ((size_t)b * H * W + (size_t)h * W + w) * C + c);
}

// window reverse + roll(+shift) + residual add into x (f32), float4 chunks
__global__ __launch_bounds__(256) void wrevadd_kernel(
    const float* __restrict__ y, float* __restrict__ x,
    int B, int H, int W, int C, int shift)
{
    size_t i = (size_t)blockIdx.x * 256 + threadIdx.x;
    int Cq = C >> 2;
    size_t total = (size_t)B * H * W * Cq;
    if (i >= total) return;
    int c = (int)(i % Cq) << 2;
    size_t r = i / Cq;
    int l = (int)(r % (H * W));
    int b = (int)(r / (H * W));
    int h = l / W, w = l % W;
    int hr = (h - shift + H) % H;
    int wr = (w - shift + W) % W;
    int Wn = W / 7, nW = (H / 7) * Wn;
    int win = b * nW + (hr / 7) * Wn + (wr / 7);
    int n = (hr % 7) * 7 + (wr % 7);
    v4f yv = *(const v4f*)(y + (size_t)(win * 49 + n) * C + c);
    v4f xv = *(v4f*)(x + i * 4);
    #pragma unroll
    for (int j = 0; j < 4; ++j) xv[j] += yv[j];
    *(v4f*)(x + i * 4) = xv;
}

// 2x2 patch-merge concat (f32 -> f32), float4 chunks
__global__ __launch_bounds__(256) void dscat_kernel(
    const float* __restrict__ x, float* __restrict__ out,
    int B, int H, int W, int C)
{
    int H2 = H / 2, W2 = W / 2, C4 = 4 * C, C4q = C4 >> 2;
    size_t i = (size_t)blockIdx.x * 256 + threadIdx.x;
    size_t total = (size_t)B * H2 * W2 * C4q;
    if (i >= total) return;
    int cq = (int)(i % C4q) << 2;
    size_t r = i / C4q;
    int q = (int)(r % W2);
    int p = (int)((r / W2) % H2);
    int b = (int)(r / ((size_t)H2 * W2));
    int g = cq / C, c = cq % C;
    int h = 2 * p + ((g == 1 || g == 3) ? 1 : 0);
    int w = 2 * q + ((g >= 2) ? 1 : 0);
    *(v4f*)(out + i * 4) =
        *(const v4f*)(x + ((size_t)b * H * W + (size_t)h * W + w) * C + c);
}

// mean over L tokens, f16 in/out with f32 accumulation
__global__ __launch_bounds__(256) void pool_kernel(
    const _Float16* __restrict__ x, _Float16* __restrict__ out,
    int B, int L, int C)
{
    int Cq = C >> 3;
    size_t i = (size_t)blockIdx.x * 256 + threadIdx.x;
    if (i >= (size_t)B * Cq) return;
    int c = (int)(i % Cq) << 3;
    int b = (int)(i / Cq);
    float s[8];
    #pragma unroll
    for (int j = 0; j < 8; ++j) s[j] = 0.0f;
    for (int n = 0; n < L; ++n) {
        v8h v = *(const v8h*)(x + ((size_t)b * L + n) * C + c);
        #pragma unroll
        for (int j = 0; j < 8; ++j) s[j] += (float)v[j];
    }
    float inv = 1.0f / (float)L;
    v8h o;
    #pragma unroll
    for (int j = 0; j < 8; ++j) o[j] = (_Float16)(s[j] * inv);
    *(v8h*)(out + (size_t)b * C + c) = o;
}

// ---------------------------------------------------------------------------
// Host orchestration
// ---------------------------------------------------------------------------
struct BlkP {
    const float *n1_g, *n1_b, *qkv_w, *qkv_b, *rpb, *proj_w, *proj_b;
    const float *n2_g, *n2_b, *fc1_w, *fc1_b, *fc2_w, *fc2_b;
};

template <typename TO>
static void launch_gemm(const _Float16* A, const float* W, const float* bias,
                        const float* resid, TO* out, int M, int N, int K,
                        int act, hipStream_t st)
{
    dim3 g((M + 63) / 64, (N + 63) / 64);
    gemm_kernel<TO><<<g, 128, 0, st>>>(A, W, bias, resid, out, M, N, K, act);
}

extern "C" void kernel_launch(void* const* d_in, const int* in_sizes, int n_in,
                              void* d_out, int out_size, void* d_ws, size_t ws_size,
                              hipStream_t stream) {
    if (n_in < 174) return;
    const int DEPTHS[4] = {2, 2, 6, 2};
    const int DIMS[4]   = {96, 192, 384, 768};
    const int HEADS[4]  = {3, 6, 12, 24};
    const int B = 64;

    // setup_inputs() insertion order: x_img, then params
    int pi = 0;
    auto F = [&](int i) { return (const float*)d_in[i]; };
    const float* x_img   = F(pi++);
    const float* patch_w = F(pi++);
    const float* patch_b = F(pi++);
    const float* pe_g    = F(pi++);
    const float* pe_b    = F(pi++);
    BlkP blk[12];
    const float *ds_g[3], *ds_b[3], *ds_w[3];
    int bi = 0;
    for (int s = 0; s < 4; ++s) {
        for (int d = 0; d < DEPTHS[s]; ++d) {
            BlkP& P = blk[bi++];
            P.n1_g = F(pi++); P.n1_b = F(pi++);
            P.qkv_w = F(pi++); P.qkv_b = F(pi++);
            P.rpb = F(pi++);
            P.proj_w = F(pi++); P.proj_b = F(pi++);
            P.n2_g = F(pi++); P.n2_b = F(pi++);
            P.fc1_w = F(pi++); P.fc1_b = F(pi++);
            P.fc2_w = F(pi++); P.fc2_b = F(pi++);
        }
        if (s < 3) { ds_g[s] = F(pi++); ds_b[s] = F(pi++); ds_w[s] = F(pi++); }
    }
    const float* norm_g = F(pi++);
    const float* norm_b = F(pi++);
    const float* head_w = F(pi++);
    const float* head_b = F(pi++);

    // workspace layout (bytes; all offsets 16B aligned)
    const size_t CAPX = 19267584;  // 200704*96 elements
    const size_t CAPQ = 57802752;  // 200704*288 elements
    const size_t CAPM = 77070336;  // 200704*384 elements
    const size_t OFF_X = 0;                       // f32 residual x
    const size_t OFF_T = OFF_X + CAPX * 4;        // f16 LN output
    const size_t OFF_W = OFF_T + CAPX * 2;        // raw: f16 windows / f32 proj out
    const size_t OFF_A = OFF_W + CAPX * 4;        // f16 attn out / ds-LN out / pooled
    const size_t OFF_Q = OFF_A + CAPX * 2;        // f16 qkv
    const size_t OFF_M = OFF_Q + CAPQ * 2;        // raw: f16 mlp hidden / f32 dscat
    const size_t need  = OFF_M + CAPM * 2;
    if (ws_size < need) return;
    char* base = (char*)d_ws;
    float*     Xb = (float*)(base + OFF_X);
    _Float16*  Tb = (_Float16*)(base + OFF_T);
    _Float16*  Wh = (_Float16*)(base + OFF_W);
    float*     Wf = (float*)(base + OFF_W);
    _Float16*  Ab = (_Float16*)(base + OFF_A);
    _Float16*  Qb = (_Float16*)(base + OFF_Q);
    _Float16*  Mh = (_Float16*)(base + OFF_M);
    float*     Mf = (float*)(base + OFF_M);

    int H = 56, W = 56;

    // patch embed + pe LayerNorm (stays f32: it IS the residual stream)
    {
        size_t tot = (size_t)B * 56 * 56 * 96;
        patch_embed_kernel<<<(unsigned)((tot + 255) / 256), 256, 0, stream>>>(
            x_img, patch_w, patch_b, Xb, B);
        int M = B * 56 * 56;
        ln_kernel<float><<<(M + 3) / 4, 128, 0, stream>>>(Xb, pe_g, pe_b, Xb, M, 96);
    }

    bi = 0;
    for (int s = 0; s < 4; ++s) {
        int C = DIMS[s], hds = HEADS[s];
        int L = H * W, M = B * L;
        int nW = (H / 7) * (W / 7);
        size_t tot8 = (size_t)M * (C >> 3);
        size_t tot4 = (size_t)M * (C >> 2);
        for (int d = 0; d < DEPTHS[s]; ++d) {
            BlkP& P = blk[bi++];
            int shift = (d % 2 == 1 && H > 7) ? 3 : 0;
            // attention branch (activation stream in f16)
            ln_kernel<_Float16><<<(M + 3) / 4, 128, 0, stream>>>(
                Xb, P.n1_g, P.n1_b, Tb, M, C);
            wpart_kernel<<<(unsigned)((tot8 + 255) / 256), 256, 0, stream>>>(
                Tb, Wh, B, H, W, C, shift);
            launch_gemm<_Float16>(Wh, P.qkv_w, P.qkv_b, nullptr, Qb, M, 3 * C, C, 0, stream);
            attn_kernel<<<B * nW * hds, 128, 0, stream>>>(
                Qb, P.rpb, Ab, nW, hds, C, H, W, shift);
            launch_gemm<float>(Ab, P.proj_w, P.proj_b, nullptr, Wf, M, C, C, 0, stream);
            wrevadd_kernel<<<(unsigned)((tot4 + 255) / 256), 256, 0, stream>>>(
                Wf, Xb, B, H, W, C, shift);
            // MLP branch
            ln_kernel<_Float16><<<(M + 3) / 4, 128, 0, stream>>>(
                Xb, P.n2_g, P.n2_b, Tb, M, C);
            launch_gemm<_Float16>(Tb, P.fc1_w, P.fc1_b, nullptr, Mh, M, 4 * C, C, 1, stream);
            launch_gemm<float>(Mh, P.fc2_w, P.fc2_b, Xb, Xb, M, C, 4 * C, 0, stream);
        }
        if (s < 3) {
            int H2 = H / 2, W2 = W / 2;
            int Md = B * H2 * W2, C4 = 4 * C;
            size_t dtot = (size_t)Md * (C4 >> 2);
            dscat_kernel<<<(unsigned)((dtot + 255) / 256), 256, 0, stream>>>(
                Xb, Mf, B, H, W, C);
            ln_kernel<_Float16><<<(Md + 3) / 4, 128, 0, stream>>>(
                Mf, ds_g[s], ds_b[s], Ab, Md, C4);
            launch_gemm<float>(Ab, ds_w[s], nullptr, nullptr, Xb, Md, 2 * C, C4, 0, stream);
            H = H2; W = W2;
        }
    }

    // final norm -> mean pool -> head
    {
        int L = H * W;           // 49
        int M = B * L;
        ln_kernel<_Float16><<<(M + 3) / 4, 128, 0, stream>>>(
            Xb, norm_g, norm_b, Tb, M, 768);
        pool_kernel<<<(B * 768 / 8 + 255) / 256, 256, 0, stream>>>(Tb, Ab, B, L, 768);
        launch_gemm<float>(Ab, head_w, head_b, nullptr, (float*)d_out, B, 4, 768, 0, stream);
    }
}
